// MTTTAttention_49950469652796
// MI455X (gfx1250) — compile-verified, matrix operand order checked by reading, not run on hardware
//
#include <hip/hip_runtime.h>
#include <hip/hip_bf16.h>

typedef float v2f  __attribute__((ext_vector_type(2)));
typedef float v8f  __attribute__((ext_vector_type(8)));
typedef __bf16 bf16;
typedef bf16 v4bf  __attribute__((ext_vector_type(4)));
typedef bf16 v8bf  __attribute__((ext_vector_type(8)));
typedef bf16 v16bf __attribute__((ext_vector_type(16)));

// fp32 exact WMMA: D = A(16x4)*B(4x16) + C
__device__ __forceinline__ v8f wmma4(v2f a, v2f b, v8f c) {
    return __builtin_amdgcn_wmma_f32_16x16x4_f32(
        false, a, false, b, (short)0, c, false, false);
}
// bf16 WMMA: D = A(16x32)*B(32x16) + C (f32 accumulate)
__device__ __forceinline__ v8f wmma32bf(v16bf a, v16bf b, v8f c) {
    return __builtin_amdgcn_wmma_f32_16x16x32_bf16(
        false, a, false, b, (short)0, c, false, false);
}

// Load one 16x32 bf16 fragment row (per ISA 7.12.2 16-bit layout):
// elements 0..7 = K[khalf..khalf+7], elements 8..15 = K[16+khalf..16+khalf+7]
__device__ __forceinline__ v16bf frag16(const bf16* rowBase, int khalf) {
    v8bf lo = *(const v8bf*)(rowBase + khalf);
    v8bf hi = *(const v8bf*)(rowBase + 16 + khalf);
    return __builtin_shufflevector(lo, hi, 0,1,2,3,4,5,6,7,8,9,10,11,12,13,14,15);
}

// ---------------------------------------------------------------------------
// Weight prep: B[K x N] fp32 -> Bt_hi/Bt_lo [N x K] bf16 (transposed split)
// ---------------------------------------------------------------------------
__global__ __launch_bounds__(256) void split_transpose_bf16(
    const float* __restrict__ B, bf16* __restrict__ Th, bf16* __restrict__ Tl,
    int K, int N)
{
    __shared__ float t[32][33];
    const int tid = threadIdx.x;
    const int k0 = blockIdx.y * 32;
    const int n0 = blockIdx.x * 32;
    #pragma unroll
    for (int i = 0; i < 4; ++i) {
        int id = tid + i * 256;
        int r = id >> 5, c = id & 31;
        t[r][c] = B[(size_t)(k0 + r) * N + n0 + c];
    }
    __syncthreads();
    #pragma unroll
    for (int i = 0; i < 4; ++i) {
        int id = tid + i * 256;
        int r = id >> 5, c = id & 31;            // r: n-local, c: k-local
        float f = t[c][r];
        bf16 h  = (bf16)f;
        float l = f - (float)h;                  // exact residual
        size_t o = (size_t)(n0 + r) * K + k0 + c;
        Th[o] = h;
        Tl[o] = (bf16)l;
    }
}

// ---------------------------------------------------------------------------
// bf16x3 split GEMM:  C[M x N] = A[M x K](f32) @ B (+ bias)
// B given pre-split & transposed: Bt_hi/Bt_lo are [N x K] bf16 row-major.
// BM=128, BN=64, KT=32, 256 threads (8 waves, 4x2 grid of 32x32 regions).
// Per K-tile per wave: 3 bf16 WMMAs per 16x16 tile (lo*hi + hi*lo + hi*hi).
// ---------------------------------------------------------------------------
#define GBM 128
#define GBN 64
#define GKT 32
#define APITCH 40   // bf16 row pitch: 80B = 20 banks -> conflict-free, 16B aligned

__global__ __launch_bounds__(256) void gemm_bf16x3(
    const float* __restrict__ A,
    const bf16* __restrict__ Bth, const bf16* __restrict__ Btl,
    const float* __restrict__ bias, float* __restrict__ Cm,
    int M, int N, int K)
{
    __shared__ __align__(16) bf16 Ash[GBM][APITCH];
    __shared__ __align__(16) bf16 Asl[GBM][APITCH];
    __shared__ __align__(16) bf16 Bsh[GBN][APITCH];
    __shared__ __align__(16) bf16 Bsl[GBN][APITCH];

    const int tid   = threadIdx.x;
    const int wave  = tid >> 5;
    const int lane  = tid & 31;
    const int lm    = lane & 15;
    const int khalf = (lane >> 4) << 3;   // 0 or 8
    const int hi    = (lane >> 4) << 3;   // C/D row offset 0 or 8

    const int wr = (wave & 3) * 32;
    const int wc = (wave >> 2) * 32;

    const long rowBase = (long)blockIdx.y * GBM;
    const long colBase = (long)blockIdx.x * GBN;

    v8f acc[2][2];
    acc[0][0] = {}; acc[0][1] = {}; acc[1][0] = {}; acc[1][1] = {};

    for (int k0 = 0; k0 < K; k0 += GKT) {
        // ---- stage A tile 128x32: split f32 -> bf16 hi/lo ----
        #pragma unroll
        for (int i = 0; i < 4; ++i) {
            int id = tid + i * 256;
            int r  = id >> 3;
            int c  = (id & 7) << 2;
            float4 f = *(const float4*)&A[(size_t)(rowBase + r) * K + k0 + c];
            v4bf h, l;
            h[0] = (bf16)f.x; l[0] = (bf16)(f.x - (float)h[0]);
            h[1] = (bf16)f.y; l[1] = (bf16)(f.y - (float)h[1]);
            h[2] = (bf16)f.z; l[2] = (bf16)(f.z - (float)h[2]);
            h[3] = (bf16)f.w; l[3] = (bf16)(f.w - (float)h[3]);
            *(v4bf*)&Ash[r][c] = h;
            *(v4bf*)&Asl[r][c] = l;
        }
        // ---- stage B tile 64x32 (pre-split bf16, pure copy) ----
        {
            int r = tid >> 2;
            int c = (tid & 3) << 3;
            size_t o = (size_t)(colBase + r) * K + k0 + c;
            *(v8bf*)&Bsh[r][c] = *(const v8bf*)&Bth[o];
            *(v8bf*)&Bsl[r][c] = *(const v8bf*)&Btl[o];
        }
        if (k0 + GKT < K)
            __builtin_prefetch(&A[(size_t)(rowBase + (tid >> 1)) * K + k0 + GKT], 0, 1);
        __syncthreads();

        // ---- fragments + 12 WMMAs (2x2 tiles x 3 products) ----
        v16bf ah0 = frag16(&Ash[wr      + lm][0], khalf);
        v16bf ah1 = frag16(&Ash[wr + 16 + lm][0], khalf);
        v16bf al0 = frag16(&Asl[wr      + lm][0], khalf);
        v16bf al1 = frag16(&Asl[wr + 16 + lm][0], khalf);
        v16bf bh0 = frag16(&Bsh[wc      + lm][0], khalf);
        v16bf bh1 = frag16(&Bsh[wc + 16 + lm][0], khalf);
        v16bf bl0 = frag16(&Bsl[wc      + lm][0], khalf);
        v16bf bl1 = frag16(&Bsl[wc + 16 + lm][0], khalf);

        acc[0][0] = wmma32bf(al0, bh0, acc[0][0]);
        acc[0][0] = wmma32bf(ah0, bl0, acc[0][0]);
        acc[0][0] = wmma32bf(ah0, bh0, acc[0][0]);
        acc[0][1] = wmma32bf(al0, bh1, acc[0][1]);
        acc[0][1] = wmma32bf(ah0, bl1, acc[0][1]);
        acc[0][1] = wmma32bf(ah0, bh1, acc[0][1]);
        acc[1][0] = wmma32bf(al1, bh0, acc[1][0]);
        acc[1][0] = wmma32bf(ah1, bl0, acc[1][0]);
        acc[1][0] = wmma32bf(ah1, bh0, acc[1][0]);
        acc[1][1] = wmma32bf(al1, bh1, acc[1][1]);
        acc[1][1] = wmma32bf(ah1, bl1, acc[1][1]);
        acc[1][1] = wmma32bf(ah1, bh1, acc[1][1]);
        __syncthreads();
    }

    #pragma unroll
    for (int i = 0; i < 2; ++i) {
        #pragma unroll
        for (int j = 0; j < 2; ++j) {
            long col  = colBase + wc + j * 16 + lm;
            float bad = bias ? bias[col] : 0.0f;
            #pragma unroll
            for (int r = 0; r < 8; ++r) {
                long row = rowBase + wr + i * 16 + hi + r;
                Cm[(size_t)row * N + col] = acc[i][j][r] + bad;
            }
        }
    }
}

// ---------------------------------------------------------------------------
// MTTT head kernel (exact fp32 WMMA): one block per (b,h), 256 threads.
//   Phase 1: A = k^T k /N, Cs = k^T v /N, mk = mean(k), mv = mean(v)
//   Phase 2: 4 GD steps:  W' = W - A W - outer(mk,b) + Cs ; b' = mv - mk W_old
//   Phase 3: out = q @ W + b  -> mid (B,N,C) layout
// ---------------------------------------------------------------------------
#define MT_C3 3072
#define MT_N  1024

__global__ __launch_bounds__(256) void mttt_head_kernel(
    const float* __restrict__ qkv, float* __restrict__ mid)
{
    __shared__ __align__(16) float smem[25536];
    float (*Ks)[68] = (float(*)[68]) &smem[0];
    float (*Vs)[68] = (float(*)[68]) &smem[32 * 68];
    float (*Qs)[68] = (float(*)[68]) &smem[0];
    float (*Am)[65] = (float(*)[65]) &smem[8704];
    float (*Cs)[65] = (float(*)[65]) &smem[8704 + 4160];
    float (*W0)[65] = (float(*)[65]) &smem[8704 + 2 * 4160];
    float (*W1)[65] = (float(*)[65]) &smem[8704 + 3 * 4160];
    float* mk   = &smem[8704 + 4 * 4160];
    float* mv   = mk + 64;
    float* bvec = mk + 128;

    const int bh = blockIdx.x;
    const int bb = bh >> 4;
    const int h  = bh & 15;
    const size_t rowBase = (size_t)bb * MT_N;
    const int qcol = h * 64;
    const int kcol = 1024 + h * 64;
    const int vcol = 2048 + h * 64;

    const int tid  = threadIdx.x;
    const int wave = tid >> 5;
    const int lane = tid & 31;
    const int lm   = lane & 15;
    const int lk   = (lane >> 4) << 1;
    const int hi   = (lane >> 4) << 3;

    // ---------------- Phase 1 ----------------
    const bool isA  = wave < 4;
    const int mbase = (wave & 3) * 16;
    v8f accT[4];
    accT[0] = {}; accT[1] = {}; accT[2] = {}; accT[3] = {};
    float msum = 0.0f;

    for (int k0 = 0; k0 < MT_N; k0 += 32) {
        #pragma unroll
        for (int i = 0; i < 2; ++i) {
            int id = tid + i * 256;
            int r  = id >> 4;
            int c  = (id & 15) << 2;
            const float* src = &qkv[(rowBase + k0 + r) * MT_C3];
            *(float4*)&Ks[r][c] = *(const float4*)&src[kcol + c];
            *(float4*)&Vs[r][c] = *(const float4*)&src[vcol + c];
        }
        __syncthreads();

        if (tid < 64) {
            #pragma unroll 8
            for (int r = 0; r < 32; ++r) msum += Ks[r][tid];
        } else if (tid < 128) {
            #pragma unroll 8
            for (int r = 0; r < 32; ++r) msum += Vs[r][tid - 64];
        }

        float (*Bsrc)[68] = isA ? Ks : Vs;
        #pragma unroll
        for (int kk = 0; kk < 32; kk += 4) {
            v2f a;
            a.x = Ks[kk + lk][mbase + lm];
            a.y = Ks[kk + lk + 1][mbase + lm];
            #pragma unroll
            for (int j = 0; j < 4; ++j) {
                v2f b;
                b.x = Bsrc[kk + lk][j * 16 + lm];
                b.y = Bsrc[kk + lk + 1][j * 16 + lm];
                accT[j] = wmma4(a, b, accT[j]);
            }
        }
        __syncthreads();
    }

    const float invN = 1.0f / (float)MT_N;
    {
        float (*Dst)[65] = isA ? Am : Cs;
        #pragma unroll
        for (int j = 0; j < 4; ++j)
            #pragma unroll
            for (int r = 0; r < 8; ++r)
                Dst[mbase + hi + r][j * 16 + lm] = accT[j][r] * invN;
    }
    if (tid < 64)       mk[tid]      = msum * invN;
    else if (tid < 128) mv[tid - 64] = msum * invN;
    for (int id = tid; id < 64 * 64; id += 256) W0[id >> 6][id & 63] = 0.0f;
    if (tid < 64) bvec[tid] = 0.0f;
    __syncthreads();

    // ---------------- Phase 2 ----------------
    float (*Wcur)[65] = W0;
    float (*Wnxt)[65] = W1;
    for (int it = 0; it < 4; ++it) {
        float bnew = 0.0f;
        if (tid < 64) {
            float s = 0.0f;
            #pragma unroll 8
            for (int i = 0; i < 64; ++i) s += mk[i] * Wcur[i][tid];
            bnew = mv[tid] - s;
        }
        #pragma unroll
        for (int j = 0; j < 2; ++j) {
            int t  = wave * 2 + j;
            int mb = (t >> 2) * 16;
            int nb = (t & 3) * 16;
            v8f acc = {};
            #pragma unroll
            for (int kk = 0; kk < 64; kk += 4) {
                v2f a, b2;
                a.x  = Am[mb + lm][kk + lk];
                a.y  = Am[mb + lm][kk + lk + 1];
                b2.x = Wcur[kk + lk][nb + lm];
                b2.y = Wcur[kk + lk + 1][nb + lm];
                acc = wmma4(a, b2, acc);
            }
            int col  = nb + lm;
            float bc = bvec[col];
            #pragma unroll
            for (int r = 0; r < 8; ++r) {
                int row = mb + hi + r;
                Wnxt[row][col] = Wcur[row][col] - acc[r] - mk[row] * bc + Cs[row][col];
            }
        }
        __syncthreads();
        if (tid < 64) bvec[tid] = bnew;
        { float (*tmp)[65] = Wcur; Wcur = Wnxt; Wnxt = tmp; }
        __syncthreads();
    }

    // ---------------- Phase 3 ----------------
    for (int c0 = 0; c0 < MT_N; c0 += 128) {
        #pragma unroll
        for (int i = 0; i < 8; ++i) {
            int id = tid + i * 256;
            int r  = id >> 4;
            int c  = (id & 15) << 2;
            *(float4*)&Qs[r][c] =
                *(const float4*)&qkv[(rowBase + c0 + r) * MT_C3 + qcol + c];
        }
        __syncthreads();

        int rb = wave * 16;
        v8f acc[4];
        acc[0] = {}; acc[1] = {}; acc[2] = {}; acc[3] = {};
        #pragma unroll
        for (int kk = 0; kk < 64; kk += 4) {
            v2f a;
            a.x = Qs[rb + lm][kk + lk];
            a.y = Qs[rb + lm][kk + lk + 1];
            #pragma unroll
            for (int j = 0; j < 4; ++j) {
                v2f b2;
                b2.x = Wcur[kk + lk][j * 16 + lm];
                b2.y = Wcur[kk + lk + 1][j * 16 + lm];
                acc[j] = wmma4(a, b2, acc[j]);
            }
        }
        #pragma unroll
        for (int j = 0; j < 4; ++j) {
            int col  = j * 16 + lm;
            float ba = bvec[col];
            #pragma unroll
            for (int r = 0; r < 8; ++r) {
                size_t row = rowBase + c0 + rb + hi + r;
                mid[row * 1024 + qcol + col] = acc[j][r] + ba;
            }
        }
        __syncthreads();
    }
}

// ---------------------------------------------------------------------------
// Launch:
//   1) split+transpose w_qkv, w_proj to bf16 hi/lo (N x K)
//   2) qkv = x @ w_qkv          (bf16x3 WMMA)
//   3) per-(b,h) MTTT           (fp32 WMMA)
//   4) out = mid @ w_proj + b   (bf16x3 WMMA)
// Workspace: qkv 96MB + mid 32MB + split weights ~16MB
// ---------------------------------------------------------------------------
extern "C" void kernel_launch(void* const* d_in, const int* in_sizes, int n_in,
                              void* d_out, int out_size, void* d_ws, size_t ws_size,
                              hipStream_t stream) {
    const float* x      = (const float*)d_in[0];
    const float* w_qkv  = (const float*)d_in[1];
    const float* w_proj = (const float*)d_in[2];
    const float* b_proj = (const float*)d_in[3];
    float* out = (float*)d_out;

    float* qkv = (float*)d_ws;                        // 8192 x 3072 f32
    float* mid = qkv + (size_t)8192 * 3072;           // 8192 x 1024 f32
    bf16*  b1h = (bf16*)(mid + (size_t)8192 * 1024);  // 3072 x 1024 bf16
    bf16*  b1l = b1h + (size_t)3072 * 1024;
    bf16*  b2h = b1l + (size_t)3072 * 1024;           // 1024 x 1024 bf16
    bf16*  b2l = b2h + (size_t)1024 * 1024;

    dim3 blk(256);
    split_transpose_bf16<<<dim3(3072 / 32, 1024 / 32), blk, 0, stream>>>(
        w_qkv, b1h, b1l, 1024, 3072);
    split_transpose_bf16<<<dim3(1024 / 32, 1024 / 32), blk, 0, stream>>>(
        w_proj, b2h, b2l, 1024, 1024);

    gemm_bf16x3<<<dim3(3072 / GBN, 8192 / GBM), blk, 0, stream>>>(
        x, b1h, b1l, nullptr, qkv, 8192, 3072, 1024);

    mttt_head_kernel<<<dim3(8 * 16), blk, 0, stream>>>(qkv, mid);

    gemm_bf16x3<<<dim3(1024 / GBN, 8192 / GBM), blk, 0, stream>>>(
        mid, b2h, b2l, b_proj, out, 8192, 1024, 1024);
}
